// _NeuralFeatures_82961588289751
// MI455X (gfx1250) — matrix operand order repaired and checked by reading, not compiled, not run on hardware
//
#include <hip/hip_runtime.h>
#include <cstdint>

// Problem constants (match reference)
#define NN   50000
#define EE   1600000
#define BB   128
#define FXc  16
#define FEc  4
#define FINc 20
#define MM   64
#define HH   32
#define NCOL (MM * HH)          // 2048 fused (m,h) columns
#define RT_PER_BLK 25           // row tiles per block (3125 / 125)
#define ROWS_PER_BLK (RT_PER_BLK * 16)

typedef __attribute__((ext_vector_type(16))) __bf16 v16bf;
typedef __attribute__((ext_vector_type(8)))  float  v8f;
typedef __attribute__((ext_vector_type(2)))  int    v2i;

#ifdef __has_builtin
#  if __has_builtin(__builtin_amdgcn_global_load_async_to_lds_b64)
#    define HAVE_ASYNC 1
#  else
#    define HAVE_ASYNC 0
#  endif
#else
#  define HAVE_ASYNC 0
#endif

#if HAVE_ASYNC
#  if __has_builtin(__builtin_amdgcn_s_wait_asynccnt)
#    define WAIT_ASYNC() __builtin_amdgcn_s_wait_asynccnt(0)
#  else
#    define WAIT_ASYNC() asm volatile("s_wait_asynccnt 0x0" ::: "memory")
#  endif
// Builtin signature (from hipcc diagnostic): (v2i32 AS1* gsrc, v2i32 AS3* ldst, Ii, Ii)
typedef __attribute__((address_space(1))) v2i as1_v2i;
typedef __attribute__((address_space(3))) v2i as3_v2i;
#endif

static __device__ __forceinline__ uint16_t f2bf(float f) {
  uint32_t u = __builtin_bit_cast(uint32_t, f);
  u += 0x7FFFu + ((u >> 16) & 1u);            // round-to-nearest-even
  return (uint16_t)(u >> 16);
}

// ---- 1) per-graph node counts ------------------------------------------------
__global__ void k_count(const int* __restrict__ batch, float* __restrict__ cnt) {
  int n = blockIdx.x * blockDim.x + threadIdx.x;
  if (n < NN) atomicAdd(&cnt[batch[n]], 1.0f);
}

__global__ void k_fact(const float* __restrict__ cnt, float* __restrict__ fact_b) {
  int b = threadIdx.x;
  if (b < BB) fact_b[b] = 1.0f / cnt[b];
}

// ---- 2) edge phase: row/col segment sums + deterministic self-loop winner ----
__global__ void k_edge1(const float* __restrict__ ea, const int* __restrict__ ei,
                        float* __restrict__ rows, float* __restrict__ cols,
                        int* __restrict__ winner) {
  int e = blockIdx.x * blockDim.x + threadIdx.x;
  if (e >= EE) return;
  int s = ei[e], d = ei[EE + e];
  float4 a = ((const float4*)ea)[e];
  atomicAdd(&rows[s * 4 + 0], a.x); atomicAdd(&rows[s * 4 + 1], a.y);
  atomicAdd(&rows[s * 4 + 2], a.z); atomicAdd(&rows[s * 4 + 3], a.w);
  atomicAdd(&cols[d * 4 + 0], a.x); atomicAdd(&cols[d * 4 + 1], a.y);
  atomicAdd(&cols[d * 4 + 2], a.z); atomicAdd(&cols[d * 4 + 3], a.w);
  if (s == d) atomicMax(&winner[s], e);
}

__global__ void k_edge2(const float* __restrict__ ea, const int* __restrict__ ei,
                        const int* __restrict__ winner, float* __restrict__ diag_e) {
  int e = blockIdx.x * blockDim.x + threadIdx.x;
  if (e >= EE) return;
  int s = ei[e], d = ei[EE + e];
  if (s == d && winner[s] == e) ((float4*)diag_e)[s] = ((const float4*)ea)[e];
}

// ---- 3) build per-node bf16 basis (K=64 padded) + batch sums -----------------
// k layout: [0..15]=x, [16..19]=diag_edge        (slot0: diag_part)
//           [20..35]=x*fn, [36..39]=rows*fn      (slot2: sum_of_rows)
//           [40..55]=x*fn, [56..59]=cols*fn      (slot3: sum_of_cols)
//           [60..63]=0 pad
__global__ void k_basis(const float* __restrict__ x, const int* __restrict__ batch,
                        const float* __restrict__ fact_b, const float* __restrict__ diag_e,
                        const float* __restrict__ rows, const float* __restrict__ cols,
                        uint16_t* __restrict__ basis,
                        float* __restrict__ sdp_acc, float* __restrict__ sa_acc) {
  int n = blockIdx.x * blockDim.x + threadIdx.x;
  if (n >= NN) return;
  int b = batch[n];
  float fb = fact_b[b];
  uint16_t* o = basis + (size_t)n * 64;
#pragma unroll
  for (int f = 0; f < 16; ++f) {
    float xv = x[n * 16 + f];
    float xf = xv * fb;
    o[f]      = f2bf(xv);
    o[20 + f] = f2bf(xf);
    o[40 + f] = f2bf(xf);
    atomicAdd(&sdp_acc[b * FINc + f], xv);   // diag_part batch sum
    atomicAdd(&sa_acc[b * FINc + f], xf);    // sum_of_cols batch sum
  }
#pragma unroll
  for (int f = 0; f < 4; ++f) {
    float de = diag_e[n * 4 + f];
    float rw = rows[n * 4 + f] * fb;
    float cl = cols[n * 4 + f] * fb;
    o[16 + f] = f2bf(de);
    o[36 + f] = f2bf(rw);
    o[56 + f] = f2bf(cl);
    atomicAdd(&sdp_acc[b * FINc + 16 + f], de);
    atomicAdd(&sa_acc[b * FINc + 16 + f], cl);
  }
  o[60] = o[61] = o[62] = o[63] = 0;
}

__global__ void k_finalize(const float* __restrict__ sdp_acc, const float* __restrict__ sa_acc,
                           const float* __restrict__ fact_b,
                           float* __restrict__ sdp, float* __restrict__ sa) {
  int i = blockIdx.x * blockDim.x + threadIdx.x;
  if (i >= BB * FINc) return;
  float fb = fact_b[i / FINc];
  sdp[i] = sdp_acc[i] * fb;          // sum_diag_part
  sa[i]  = sa_acc[i] * fb * fb;      // sum_all (sum_of_cols already has one fact_n)
}

// ---- 4) pack weights (slots 0,2,3) into bf16 B matrix, col-major K=64 --------
__global__ void k_bpack(const float* __restrict__ ke, uint16_t* __restrict__ Bp) {
  int col = blockIdx.x * blockDim.x + threadIdx.x;
  if (col >= NCOL) return;
  int m = col >> 5, h = col & 31;
  uint16_t* o = Bp + (size_t)col * 64;
#pragma unroll
  for (int f = 0; f < FINc; ++f) {
    o[f]      = f2bf(ke[((m * 5 + 0) * HH + h) * FINc + f]);
    o[20 + f] = f2bf(ke[((m * 5 + 2) * HH + h) * FINc + f]);
    o[40 + f] = f2bf(ke[((m * 5 + 3) * HH + h) * FINc + f]);
  }
  o[60] = o[61] = o[62] = o[63] = 0;
}

// ---- 5) per-(graph, column) offset: batch slots 1,4 + bias_equiv -------------
__global__ void k_coff(const float* __restrict__ ke, const float* __restrict__ be,
                       const float* __restrict__ sdp, const float* __restrict__ sa,
                       float* __restrict__ Coff) {
  int idx = blockIdx.x * blockDim.x + threadIdx.x;
  if (idx >= BB * NCOL) return;
  int b = idx / NCOL, col = idx % NCOL;
  int m = col >> 5, h = col & 31;
  float s = be[m * HH + h];
  const float* k1 = ke + ((m * 5 + 1) * HH + h) * FINc;
  const float* k4 = ke + ((m * 5 + 4) * HH + h) * FINc;
#pragma unroll
  for (int f = 0; f < FINc; ++f)
    s += sdp[b * FINc + f] * k1[f] + sa[b * FINc + f] * k4[f];
  Coff[idx] = s;
}

// ---- A-tile staging: 16 rows x 64 k (2KB), 256 threads x 8B ------------------
#if HAVE_ASYNC
static __device__ __forceinline__ void stage_tile(const uint16_t* basis, int rowbase,
                                                  uint16_t* dst, int tid) {
  // ASYNCcnt-tracked LDS DMA (global_load_async_to_lds_b64)
  as1_v2i* g = (as1_v2i*)(uintptr_t)((const char*)basis + (size_t)rowbase * 128 +
                                     (size_t)tid * 8);
  as3_v2i* l = (as3_v2i*)(uintptr_t)(dst + (size_t)tid * 4);   // low 32b = LDS offset
  __builtin_amdgcn_global_load_async_to_lds_b64(g, l, 0, 0);
}
#else
static __device__ __forceinline__ void stage_tile(const uint16_t* basis, int rowbase,
                                                  uint16_t* dst, int tid) {
  ((uint64_t*)dst)[tid] = ((const uint64_t*)(basis + (size_t)rowbase * 64))[tid];
}
#endif

// ---- 6) fused WMMA GEMM + relu + *kinv + per-graph reduction -----------------
// grid = (125 row blocks of 400 rows, 16 col groups), block = 256 (8 waves).
// Each wave: one 16-col tile, B resident in regs, 25 tiles -> 50 WMMAs.
// Per-lane column accumulator flushed only on graph-ID change (batch sorted).
__global__ __launch_bounds__(256) void k_gemm(
    const uint16_t* __restrict__ basis, const uint16_t* __restrict__ Bp,
    const float* __restrict__ Coff, const float* __restrict__ kinv,
    const int* __restrict__ batch, float* __restrict__ psi_acc) {
  __shared__ uint16_t sA[2][16 * 64];
  __shared__ int sBatch[ROWS_PER_BLK];

  const int tid = threadIdx.x;
  const int rowstart = blockIdx.x * ROWS_PER_BLK;
  for (int i = tid; i < ROWS_PER_BLK; i += 256) sBatch[i] = batch[rowstart + i];

  const int lane = tid & 31;
  const int wave = tid >> 5;
  const int colbase = (blockIdx.y * 8 + wave) * 16;
  const int col  = colbase + (lane & 15);
  const int half = lane >> 4;
  const int arow = lane & 15;
  const int m = colbase >> 5;                 // all 16 cols share one m (H=32)
  const float kv = kinv[col];

  union U { v16bf v; uint64_t q[4]; };
  U B0, B1;                                   // B operand resident across all tiles
  {
    const uint64_t* bp0 = (const uint64_t*)(Bp + (size_t)col * 64 + 8 * half);
    const uint64_t* bp1 = (const uint64_t*)(Bp + (size_t)col * 64 + 32 + 8 * half);
    B0.q[0] = bp0[0]; B0.q[1] = bp0[1]; B0.q[2] = bp0[4]; B0.q[3] = bp0[5];
    B1.q[0] = bp1[0]; B1.q[1] = bp1[1]; B1.q[2] = bp1[4]; B1.q[3] = bp1[5];
  }

  stage_tile(basis, rowstart, &sA[0][0], tid);
#if HAVE_ASYNC
  WAIT_ASYNC();
#endif
  __syncthreads();

  auto flush = [&](float v, int b) {          // column-reduce within 16-lane half
    v += __shfl_xor(v, 1, 32);
    v += __shfl_xor(v, 2, 32);
    v += __shfl_xor(v, 4, 32);
    v += __shfl_xor(v, 8, 32);
    if ((lane & 15) == 0) atomicAdd(&psi_acc[b * MM + m], v);
  };

  int   bcur  = sBatch[8 * half];             // first row this half processes
  float accv  = 0.0f;
  float coffv = Coff[(size_t)bcur * NCOL + col];

  int buf = 0;
  for (int t = 0; t < RT_PER_BLK; ++t) {
    if (t + 1 < RT_PER_BLK)
      stage_tile(basis, rowstart + (t + 1) * 16, &sA[buf ^ 1][0], tid);

    // A 16x32 bf16 layout: lane half holds K = 8h+[0..7] then 16+8h+[0..7]
    U A0, A1;
    const uint16_t* a = &sA[buf][arow * 64];
    const uint64_t* ap;
    ap = (const uint64_t*)(a +       8 * half); A0.q[0] = ap[0]; A0.q[1] = ap[1];
    ap = (const uint64_t*)(a + 16 +  8 * half); A0.q[2] = ap[0]; A0.q[3] = ap[1];
    ap = (const uint64_t*)(a + 32 +  8 * half); A1.q[0] = ap[0]; A1.q[1] = ap[1];
    ap = (const uint64_t*)(a + 48 +  8 * half); A1.q[2] = ap[0]; A1.q[3] = ap[1];

    v8f acc = {};
    acc = __builtin_amdgcn_wmma_f32_16x16x32_bf16(false, A0.v, false, B0.v,
                                                  (short)0, acc, false, false);
    acc = __builtin_amdgcn_wmma_f32_16x16x32_bf16(false, A1.v, false, B1.v,
                                                  (short)0, acc, false, false);

    // rows handled by this half in tile t: rb .. rb+7 (uniform across the half)
    const int rb = t * 16 + 8 * half;
    const int bfirst = sBatch[rb], blast = sBatch[rb + 7];
    if (bfirst == blast) {                    // common: whole run in one graph
      if (bfirst != bcur) {
        flush(accv, bcur);
        bcur = bfirst; accv = 0.0f;
        coffv = Coff[(size_t)bcur * NCOL + col];
      }
#pragma unroll
      for (int i = 0; i < 8; ++i) {
        float v = acc[i] + coffv;
        accv += v > 0.0f ? v * kv : 0.0f;
      }
    } else {                                  // rare: graph boundary inside run
#pragma unroll
      for (int i = 0; i < 8; ++i) {
        int b = sBatch[rb + i];
        if (b != bcur) {
          flush(accv, bcur);
          bcur = b; accv = 0.0f;
          coffv = Coff[(size_t)bcur * NCOL + col];
        }
        float v = acc[i] + coffv;
        accv += v > 0.0f ? v * kv : 0.0f;
      }
    }

    if (t + 1 < RT_PER_BLK) {
#if HAVE_ASYNC
      WAIT_ASYNC();
#endif
      __syncthreads();
      buf ^= 1;
    }
  }
  flush(accv, bcur);
}

// ---- 7) output: scale by fact_b, subtract zerograph (bias_inv cancels) -------
__global__ void k_final(const float* __restrict__ psi_acc, const float* __restrict__ fact_b,
                        const float* __restrict__ be, const float* __restrict__ kinv,
                        float* __restrict__ out) {
  int idx = blockIdx.x * blockDim.x + threadIdx.x;
  if (idx >= BB * MM) return;
  int b = idx / MM, m = idx % MM;
  float zg = 0.0f;
#pragma unroll
  for (int h = 0; h < HH; ++h) {
    float v = be[m * HH + h];
    if (v > 0.0f) zg += v * kinv[m * HH + h];
  }
  out[idx] = psi_acc[idx] * fact_b[b] - zg;
}

extern "C" void kernel_launch(void* const* d_in, const int* in_sizes, int n_in,
                              void* d_out, int out_size, void* d_ws, size_t ws_size,
                              hipStream_t stream) {
  (void)in_sizes; (void)n_in; (void)out_size; (void)ws_size;
  const float* x    = (const float*)d_in[0];   // (N,16)
  const float* ea   = (const float*)d_in[1];   // (E,4)
  const float* ke   = (const float*)d_in[2];   // (M,5,H,FIN)
  const float* kinv = (const float*)d_in[3];   // (M,1,1,H) -> flat m*H+h
  const float* be   = (const float*)d_in[4];   // (1,M,1,H) -> flat m*H+h
  // d_in[5] = bias_inv: cancels in psi - zerograph
  const int* ei     = (const int*)d_in[6];     // (2,E)
  const int* batch  = (const int*)d_in[7];     // (N,)
  float* out = (float*)d_out;

  // Workspace layout (~10.4 MB). Leading region is zero-initialized each call.
  float* w = (float*)d_ws;
  float* cnt     = w; w += BB;
  float* sdp_acc = w; w += BB * FINc;
  float* sa_acc  = w; w += BB * FINc;
  float* psi_acc = w; w += BB * MM;
  float* diag_e  = w; w += NN * FEc;           // 16B aligned (float4 stores)
  float* rowsv   = w; w += NN * FEc;
  float* colsv   = w; w += NN * FEc;
  int* winner    = (int*)w; w += NN;
  size_t zero_bytes = (size_t)((char*)w - (char*)d_ws);
  float* fact_b = w; w += BB;
  float* sdp    = w; w += BB * FINc;
  float* sa     = w; w += BB * FINc;
  float* Coff   = w; w += BB * NCOL;
  uint16_t* basis = (uint16_t*)w;              // N x 64 bf16
  uint16_t* Bp    = basis + (size_t)NN * 64;   // 2048 x 64 bf16

  (void)hipMemsetAsync(d_ws, 0, zero_bytes, stream);

  k_count   <<<(NN + 255) / 256, 256, 0, stream>>>(batch, cnt);
  k_fact    <<<1, BB, 0, stream>>>(cnt, fact_b);
  k_edge1   <<<(EE + 255) / 256, 256, 0, stream>>>(ea, ei, rowsv, colsv, winner);
  k_edge2   <<<(EE + 255) / 256, 256, 0, stream>>>(ea, ei, winner, diag_e);
  k_basis   <<<(NN + 255) / 256, 256, 0, stream>>>(x, batch, fact_b, diag_e, rowsv,
                                                   colsv, basis, sdp_acc, sa_acc);
  k_finalize<<<(BB * FINc + 255) / 256, 256, 0, stream>>>(sdp_acc, sa_acc, fact_b, sdp, sa);
  k_bpack   <<<(NCOL + 255) / 256, 256, 0, stream>>>(ke, Bp);
  k_coff    <<<(BB * NCOL + 255) / 256, 256, 0, stream>>>(ke, be, sdp, sa, Coff);

  dim3 g(NN / ROWS_PER_BLK, NCOL / (16 * 8));  // (125, 16), exact tiling
  k_gemm    <<<g, 256, 0, stream>>>(basis, Bp, Coff, kinv, batch, psi_acc);
  k_final   <<<(BB * MM + 255) / 256, 256, 0, stream>>>(psi_acc, fact_b, be, kinv, out);
}
